// Hypernetwork_3874060501312
// MI455X (gfx1250) — matrix operand order repaired and checked by reading, not compiled, not run on hardware
//
#include <hip/hip_runtime.h>
#include <hip/hip_bf16.h>

// ---------------------------------------------------------------------------
// Hypernetwork + per-mask batched MLP for MI455X (gfx1250, wave32, WMMA).
//
// Roofline: the per-mask MLP is ~45 GFLOP over L2-resident data (4MB bf16
// data + 22MB bf16 weights << 192MB L2) -> matrix-core bound. Use
// v_wmma_f32_16x16x32_bf16 with fp32 accumulation. Weights are staged
// LDS-side via the Tensor Data Mover (TENSOR_LOAD_TO_LDS / TENSORcnt, one
// DMA op per block), biases via GLOBAL_LOAD_ASYNC_TO_LDS_B128 (ASYNCcnt),
// both overlapped with the VGPR-path gather of the (B, sel) tile.
// ---------------------------------------------------------------------------

typedef __attribute__((ext_vector_type(16))) __bf16 v16bf;
typedef __attribute__((ext_vector_type(8)))  __bf16 v8bf;
typedef __attribute__((ext_vector_type(8)))  float  v8f;
typedef __attribute__((ext_vector_type(4)))  unsigned int u32x4;
typedef __attribute__((ext_vector_type(4)))  int          i32x4;
typedef __attribute__((ext_vector_type(8)))  int          i32x8;

typedef __attribute__((address_space(3))) char  lds_char_t;
// async-to-LDS builtin params: v4i32 in AS1 (HIP __device__) / AS3 (__shared__)
typedef __attribute__((address_space(1))) i32x4 g_v4i_t;
typedef __attribute__((address_space(3))) i32x4 l_v4i_t;

#if __has_builtin(__builtin_amdgcn_global_load_async_to_lds_b128)
#define HAVE_ASYNC 1
#else
#define HAVE_ASYNC 0
#endif
#if __has_builtin(__builtin_amdgcn_tensor_load_to_lds)
#if (__clang_major__ >= 23)
#define HAVE_TDM6 1   // 6-arg form (clang-23 / amdgpu-toolchain lane)
#define HAVE_TDM5 0
#else
#define HAVE_TDM6 0
#define HAVE_TDM5 1   // 5-arg form (ROCm 7.2 / clang-22)
#endif
#else
#define HAVE_TDM6 0
#define HAVE_TDM5 0
#endif
#define HAVE_TDM (HAVE_TDM5 || HAVE_TDM6)

// Problem sizes
constexpr int Dd   = 1024;
constexpr int Mm   = 256;
constexpr int Bb   = 2048;
constexpr int Pp   = 43232;
constexpr int OUTN = 32;

// Offsets inside one mask's flattened weight vector w[m] (element units)
constexpr int OFF_W1 = 0;       // 128 x 256
constexpr int OFF_B1 = 32768;   // 128
constexpr int OFF_W2 = 32896;   // 64 x 128
constexpr int OFF_B2 = 41088;   // 64
constexpr int OFF_W3 = 41152;   // 32 x 64
constexpr int OFF_B3 = 43200;   // 32

// Tiling
constexpr int BT = 128;   // batch rows per block
constexpr int MG = 8;     // masks per block (looped; layer-3 acc persists)

// Dynamic LDS layout (bytes) -- total ~197KB, fits CDNA5's 320KB/WGP
constexpr int SM_X   = 0;        // BT*256 bf16          = 65536
constexpr int SM_W1  = 65536;    // 128*256 bf16         = 65536
constexpr int SM_H1  = 131072;   // BT*128 bf16          = 32768
constexpr int SM_W2  = 163840;   // 64*128 bf16          = 16384
constexpr int SM_H2  = 180224;   // BT*64 bf16           = 16384
constexpr int SM_W3  = 196608;   // 32*64 bf16           = 4096
constexpr int SM_BB  = 200704;   // 224 bf16 biases (b1|b2|b3) = 448
constexpr int SMEM_BYTES = 201216;

// ---------------------------------------------------------------------------
// Device helpers for CDNA5 async staging
// ---------------------------------------------------------------------------
__device__ __forceinline__ unsigned lds_addr32(void* p) {
  return (unsigned)(size_t)(lds_char_t*)p;
}

#if HAVE_TDM
// One 1-D contiguous TDM transfer: nelem bf16 elements global -> LDS.
// D# layout per CDNA5 ISA (count=1, type=2,
// data_size=1 (2B), tensor_dim0=tile_dim0=nelem, dim0 stride=nelem,
// tensor_dim1=tile_dim1=1, remaining groups zero).
__device__ __forceinline__ void tdm_copy_1d(unsigned lds_addr, const void* gsrc,
                                            unsigned nelem) {
  unsigned long long ga = (unsigned long long)(size_t)gsrc;
  u32x4 g0;
  g0[0] = 1u;                                   // count=1, user flags 0
  g0[1] = lds_addr;                             // lds_addr [63:32]
  g0[2] = (unsigned)(ga & 0xffffffffu);         // global_addr [95:64]
  g0[3] = (unsigned)((ga >> 32) & 0x1ffffffu)   // global_addr [120:96]
          | (2u << 30);                         // type=2 ("image") [127:126]
  i32x8 g1;
  g1[0] = (int)(1u << 16);                      // wg_mask=0, data_size=1 (2B)
  g1[1] = (int)((nelem & 0xffffu) << 16);       // tensor_dim0[15:0] @ [63:48]
  g1[2] = (int)((nelem >> 16) & 0xffffu)        // tensor_dim0[31:16] @ [79:64]
          | (int)(1u << 16);                    // tensor_dim1=1 @ [111:80]
  g1[3] = (int)((nelem & 0xffffu) << 16);       // tile_dim0 @ [127:112]
  g1[4] = 1;                                    // tile_dim1=1, tile_dim2=0
  g1[5] = (int)nelem;                           // tensor_dim0_stride lo32
  g1[6] = 0;                                    // stride hi16 | dim1_stride lo
  g1[7] = 0;
  i32x4 gz4 = {0, 0, 0, 0};
#if HAVE_TDM6
  i32x8 gz8 = {0, 0, 0, 0, 0, 0, 0, 0};
  __builtin_amdgcn_tensor_load_to_lds(g0, g1, gz4, gz4, gz8, 0);
#else
  __builtin_amdgcn_tensor_load_to_lds(g0, g1, gz4, gz4, 0);
#endif
}
#endif

#if HAVE_ASYNC
__device__ __forceinline__ void async_cp_b128(void* ldst, const void* gsrc,
                                              int bytes, int tid) {
  l_v4i_t* d = (l_v4i_t*)ldst;
  g_v4i_t* s = (g_v4i_t*)gsrc;
  for (int i = tid; i < bytes / 16; i += 256)
    __builtin_amdgcn_global_load_async_to_lds_b128(s + i, d + i, 0, 0);
}
#endif

// ---------------------------------------------------------------------------
// Utility kernels
// ---------------------------------------------------------------------------
__global__ void zero_out_kernel(float* __restrict__ p, int n) {
  int i = blockIdx.x * blockDim.x + threadIdx.x;
  if (i < n) p[i] = 0.0f;
}

__global__ void data_to_bf16_kernel(const float* __restrict__ src,
                                    unsigned short* __restrict__ dst, int n) {
  int i = blockIdx.x * blockDim.x + threadIdx.x;
  if (i < n) ((__bf16*)dst)[i] = (__bf16)src[i];
}

// sel[m][k] = ascending indices where mask[m][j] == 1 (matches top_k on 0/1
// mask with lowest-index tie-breaking). One thread per mask row.
__global__ void sel_kernel(const float* __restrict__ mask, int* __restrict__ sel) {
  int m = blockIdx.x * blockDim.x + threadIdx.x;
  if (m >= Mm) return;
  const float* mr = mask + (size_t)m * Dd;
  int cnt = 0;
  for (int j = 0; j < Dd; ++j) {
    if (mr[j] > 0.5f) {
      if (cnt < 256) sel[m * 256 + cnt] = j;
      ++cnt;
    }
  }
}

// h3[m][0..63]: the small hypernetwork trunk. One block per mask, 128 thr.
__global__ __launch_bounds__(128)
void hyper_hidden_kernel(const float* __restrict__ mask,
                         const float* __restrict__ Wh1, const float* __restrict__ bh1,
                         const float* __restrict__ Wh2, const float* __restrict__ bh2,
                         const float* __restrict__ Wh3, const float* __restrict__ bh3,
                         float* __restrict__ h3out) {
  __shared__ float mk[Dd];
  __shared__ float h1[128];
  __shared__ float h2[64];
  const int m = blockIdx.x;
  const int tid = threadIdx.x;
  for (int i = tid; i < Dd; i += 128) mk[i] = mask[(size_t)m * Dd + i];
  __syncthreads();
  {
    float s = bh1[tid];
    const float* wr = Wh1 + (size_t)tid * Dd;
    for (int i = 0; i < Dd; ++i) s += mk[i] * wr[i];
    h1[tid] = s > 0.0f ? s : 0.0f;
  }
  __syncthreads();
  if (tid < 64) {
    float s = bh2[tid];
    const float* wr = Wh2 + (size_t)tid * 128;
    for (int i = 0; i < 128; ++i) s += h1[i] * wr[i];
    h2[tid] = s > 0.0f ? s : 0.0f;
  }
  __syncthreads();
  if (tid < 64) {
    float s = bh3[tid];
    const float* wr = Wh3 + (size_t)tid * 64;
    for (int i = 0; i < 64; ++i) s += h2[i] * wr[i];
    h3out[(size_t)m * 64 + tid] = s;
  }
}

// w[m][p] = dot(h3[m], Wout[p]) + bout[p], stored bf16 (weights + biases).
__global__ __launch_bounds__(256)
void hyper_wout_kernel(const float* __restrict__ h3,
                       const float* __restrict__ Wout,
                       const float* __restrict__ bout,
                       unsigned short* __restrict__ wbf_u) {
  __shared__ float h[64];
  const int m = blockIdx.y;
  const int p = blockIdx.x * 256 + threadIdx.x;
  if (threadIdx.x < 64) h[threadIdx.x] = h3[(size_t)m * 64 + threadIdx.x];
  __syncthreads();
  if (p < Pp) {
    float s = bout[p];
    const float* wr = Wout + (size_t)p * 64;
#pragma unroll 8
    for (int i = 0; i < 64; ++i) s += h[i] * wr[i];
    __bf16* wbf = (__bf16*)wbf_u;
    wbf[(size_t)m * Pp + p] = (__bf16)s;
  }
}

// ---------------------------------------------------------------------------
// Main kernel: gathered batched MLP, mean over masks, via bf16 WMMA.
// grid = (Bb/BT, Mm/MG), 256 threads (8 wave32).
// Fragment layouts used here (A and B share contiguous k = 16h + s):
//   A (16x32 bf16): lane L holds row M=L%16; K = kk*32 + (L>=16?16:0) + {0..15}
//                   (two adjacent 16B chunks)
//   B (32x16 bf16): lane L holds col N=L%16; K = kk*32 + (L>=16?16:0) + {0..15}
//                   (one contiguous 32B chunk)
//   C/D (16x16 f32): vgpr r, lane l -> row = r + 8*(l>>4), col = l&15
// ---------------------------------------------------------------------------
__global__ __launch_bounds__(256, 1)
void hyper_mlp_main(const unsigned short* __restrict__ databf_u,
                    const int* __restrict__ sel,
                    const unsigned short* __restrict__ wbf_u,
                    float* __restrict__ out) {
  extern __shared__ char smem[];
  __bf16* Xs  = (__bf16*)(smem + SM_X);
  __bf16* W1s = (__bf16*)(smem + SM_W1);
  __bf16* H1s = (__bf16*)(smem + SM_H1);
  __bf16* W2s = (__bf16*)(smem + SM_W2);
  __bf16* H2s = (__bf16*)(smem + SM_H2);
  __bf16* W3s = (__bf16*)(smem + SM_W3);
  __bf16* Bbs = (__bf16*)(smem + SM_BB);   // b1[0..127] | b2[128..191] | b3[192..223]

  const __bf16* databf = (const __bf16*)databf_u;
  const __bf16* wbf    = (const __bf16*)wbf_u;
  const int tid  = threadIdx.x;
  const int wave = tid >> 5;
  const int lane = tid & 31;
  const int lh   = lane >> 4;   // lane half (0/1)
  const int ll   = lane & 15;
  const int b0   = blockIdx.x * BT;
  const int m0   = blockIdx.y * MG;
  const int kselA = lh ? 16 : 0;
  const int kselB = lh ? 16 : 0;

  v8f acc3[2];
  float bsum[2] = {0.0f, 0.0f};
#pragma unroll
  for (int t = 0; t < 2; ++t) {
#pragma unroll
    for (int e = 0; e < 8; ++e) acc3[t][e] = 0.0f;
  }

  for (int mi = 0; mi < MG; ++mi) {
    const int m = m0 + mi;
    const __bf16* wm = wbf + (size_t)m * Pp;

    // ---- stage weights into LDS: TDM for W1/W2/W3, ASYNC for biases -------
#if HAVE_TDM
    if (wave == 0) {
      tdm_copy_1d(lds_addr32(W1s), wm + OFF_W1, 32768u);
      tdm_copy_1d(lds_addr32(W2s), wm + OFF_W2, 8192u);
      tdm_copy_1d(lds_addr32(W3s), wm + OFF_W3, 2048u);
    }
#elif HAVE_ASYNC
    async_cp_b128(W1s, wm + OFF_W1, 65536, tid);
    async_cp_b128(W2s, wm + OFF_W2, 16384, tid);
    async_cp_b128(W3s, wm + OFF_W3, 4096, tid);
#else
    {
      const uint4* s1 = (const uint4*)(wm + OFF_W1);
      uint4* d1 = (uint4*)W1s;
      for (int i = tid; i < 65536 / 16; i += 256) d1[i] = s1[i];
      const uint4* s2 = (const uint4*)(wm + OFF_W2);
      uint4* d2 = (uint4*)W2s;
      for (int i = tid; i < 16384 / 16; i += 256) d2[i] = s2[i];
      const uint4* s3 = (const uint4*)(wm + OFF_W3);
      uint4* d3 = (uint4*)W3s;
      for (int i = tid; i < 4096 / 16; i += 256) d3[i] = s3[i];
    }
#endif
#if HAVE_ASYNC
    async_cp_b128(Bbs,       wm + OFF_B1, 256, tid);   // b1 (128 bf16)
    async_cp_b128(Bbs + 128, wm + OFF_B2, 128, tid);   // b2 (64 bf16)
    async_cp_b128(Bbs + 192, wm + OFF_B3,  64, tid);   // b3 (32 bf16)
#else
    if (tid < 128) Bbs[tid]       = wm[OFF_B1 + tid];
    if (tid < 64)  Bbs[128 + tid] = wm[OFF_B2 + tid];
    if (tid < 32)  Bbs[192 + tid] = wm[OFF_B3 + tid];
#endif

    // ---- gather X tile while the DMAs run: Xs[r][tid] = databf[b0+r][col] --
    {
      const int col = sel[m * 256 + tid];            // loop-invariant per thread
      const __bf16* dcol = databf + col;
#pragma unroll 8
      for (int r = 0; r < BT; ++r)
        Xs[r * 256 + tid] = dcol[(size_t)(b0 + r) * Dd];
    }
    if (mi + 1 < MG) __builtin_prefetch(wbf + (size_t)(m + 1) * Pp, 0, 0);

#if HAVE_TDM
    if (wave == 0) __builtin_amdgcn_s_wait_tensorcnt(0);
#endif
#if HAVE_ASYNC
#if __has_builtin(__builtin_amdgcn_s_wait_asynccnt)
    __builtin_amdgcn_s_wait_asynccnt(0);
#else
    asm volatile("s_wait_asynccnt 0x0" ::: "memory");
#endif
#endif
    __syncthreads();

    // ---- layer 1: H1(BTx128) = relu(X(BTx256) @ W1^T + b1), K=256 ----
    {
      v8f acc[8];
#pragma unroll
      for (int t = 0; t < 8; ++t) {
#pragma unroll
        for (int e = 0; e < 8; ++e) acc[t][e] = 0.0f;
      }
      const int rowA = wave * 16 + ll;
      for (int kk = 0; kk < 8; ++kk) {
        const int kb = kk * 32 + kselA;
        v8bf alo = *(const v8bf*)(Xs + rowA * 256 + kb);
        v8bf ahi = *(const v8bf*)(Xs + rowA * 256 + kb + 8);
        v16bf a = __builtin_shufflevector(alo, ahi, 0, 1, 2, 3, 4, 5, 6, 7,
                                          8, 9, 10, 11, 12, 13, 14, 15);
        const int kbB = kk * 32 + kselB;
#pragma unroll
        for (int tn = 0; tn < 8; ++tn) {
          v16bf b = *(const v16bf*)(W1s + (tn * 16 + ll) * 256 + kbB);
          acc[tn] = __builtin_amdgcn_wmma_f32_16x16x32_bf16(
              false, a, false, b, (short)0, acc[tn], false, false);
        }
      }
#pragma unroll
      for (int tn = 0; tn < 8; ++tn) {
        const int o = tn * 16 + ll;
        const float bias = (float)Bbs[o];
#pragma unroll
        for (int r = 0; r < 8; ++r) {
          float v = acc[tn][r] + bias;
          v = v > 0.0f ? v : 0.0f;
          H1s[(wave * 16 + r + 8 * lh) * 128 + o] = (__bf16)v;
        }
      }
    }
    __syncthreads();

    // ---- layer 2: H2(BTx64) = relu(H1 @ W2^T + b2), K=128 ----
    {
      v8f acc[4];
#pragma unroll
      for (int t = 0; t < 4; ++t) {
#pragma unroll
        for (int e = 0; e < 8; ++e) acc[t][e] = 0.0f;
      }
      const int rowA = wave * 16 + ll;
      for (int kk = 0; kk < 4; ++kk) {
        const int kb = kk * 32 + kselA;
        v8bf alo = *(const v8bf*)(H1s + rowA * 128 + kb);
        v8bf ahi = *(const v8bf*)(H1s + rowA * 128 + kb + 8);
        v16bf a = __builtin_shufflevector(alo, ahi, 0, 1, 2, 3, 4, 5, 6, 7,
                                          8, 9, 10, 11, 12, 13, 14, 15);
        const int kbB = kk * 32 + kselB;
#pragma unroll
        for (int tn = 0; tn < 4; ++tn) {
          v16bf b = *(const v16bf*)(W2s + (tn * 16 + ll) * 128 + kbB);
          acc[tn] = __builtin_amdgcn_wmma_f32_16x16x32_bf16(
              false, a, false, b, (short)0, acc[tn], false, false);
        }
      }
#pragma unroll
      for (int tn = 0; tn < 4; ++tn) {
        const int o = tn * 16 + ll;
        const float bias = (float)Bbs[128 + o];
#pragma unroll
        for (int r = 0; r < 8; ++r) {
          float v = acc[tn][r] + bias;
          v = v > 0.0f ? v : 0.0f;
          H2s[(wave * 16 + r + 8 * lh) * 64 + o] = (__bf16)v;
        }
      }
    }
    __syncthreads();

    // ---- layer 3: acc3 += H2 @ W3^T  (bias accumulated separately), K=64 ----
    {
      const int rowA = wave * 16 + ll;
      for (int kk = 0; kk < 2; ++kk) {
        const int kb = kk * 32 + kselA;
        v8bf alo = *(const v8bf*)(H2s + rowA * 64 + kb);
        v8bf ahi = *(const v8bf*)(H2s + rowA * 64 + kb + 8);
        v16bf a = __builtin_shufflevector(alo, ahi, 0, 1, 2, 3, 4, 5, 6, 7,
                                          8, 9, 10, 11, 12, 13, 14, 15);
        const int kbB = kk * 32 + kselB;
#pragma unroll
        for (int tn = 0; tn < 2; ++tn) {
          v16bf b = *(const v16bf*)(W3s + (tn * 16 + ll) * 64 + kbB);
          acc3[tn] = __builtin_amdgcn_wmma_f32_16x16x32_bf16(
              false, a, false, b, (short)0, acc3[tn], false, false);
        }
      }
#pragma unroll
      for (int tn = 0; tn < 2; ++tn) bsum[tn] += (float)Bbs[192 + tn * 16 + ll];
    }
    __syncthreads();  // protect LDS before next mask's staging
  }

  // ---- epilogue: out[b][o] += (acc + sum_m b3[m][o]) / M  (f32 atomics) ----
  const float scale = 1.0f / (float)Mm;
#pragma unroll
  for (int tn = 0; tn < 2; ++tn) {
    const int col = tn * 16 + ll;
#pragma unroll
    for (int r = 0; r < 8; ++r) {
      const int row = b0 + wave * 16 + r + 8 * lh;
      unsafeAtomicAdd(&out[row * OUTN + col], (acc3[tn][r] + bsum[tn]) * scale);
    }
  }
}

// ---------------------------------------------------------------------------
// Launch
// ---------------------------------------------------------------------------
extern "C" void kernel_launch(void* const* d_in, const int* in_sizes, int n_in,
                              void* d_out, int out_size, void* d_ws, size_t ws_size,
                              hipStream_t stream) {
  const float* data = (const float*)d_in[0];
  const float* mask = (const float*)d_in[1];
  const float* Wh1  = (const float*)d_in[2];
  const float* bh1  = (const float*)d_in[3];
  const float* Wh2  = (const float*)d_in[4];
  const float* bh2  = (const float*)d_in[5];
  const float* Wh3  = (const float*)d_in[6];
  const float* bh3  = (const float*)d_in[7];
  const float* Wout = (const float*)d_in[8];
  const float* bout = (const float*)d_in[9];
  float* out = (float*)d_out;

  // Workspace: bf16 weights (22.1MB) | h3 (64KB) | sel (256KB) | bf16 data (4MB)
  char* ws = (char*)d_ws;
  unsigned short* wbf = (unsigned short*)ws;
  size_t wbfBytes = (((size_t)Mm * Pp * 2) + 255) & ~(size_t)255;
  float* h3 = (float*)(ws + wbfBytes);
  size_t h3Bytes = (((size_t)Mm * 64 * 4) + 255) & ~(size_t)255;
  int* sel = (int*)(ws + wbfBytes + h3Bytes);
  size_t selBytes = (((size_t)Mm * 256 * 4) + 255) & ~(size_t)255;
  unsigned short* databf = (unsigned short*)(ws + wbfBytes + h3Bytes + selBytes);

  zero_out_kernel<<<dim3(64), dim3(1024), 0, stream>>>(out, Bb * OUTN);
  data_to_bf16_kernel<<<dim3((Bb * Dd) / 1024), dim3(1024), 0, stream>>>(
      data, databf, Bb * Dd);
  sel_kernel<<<dim3(1), dim3(256), 0, stream>>>(mask, sel);
  hyper_hidden_kernel<<<dim3(Mm), dim3(128), 0, stream>>>(mask, Wh1, bh1, Wh2,
                                                          bh2, Wh3, bh3, h3);
  hyper_wout_kernel<<<dim3((Pp + 255) / 256, Mm), dim3(256), 0, stream>>>(
      h3, Wout, bout, wbf);
  hyper_mlp_main<<<dim3(Bb / BT, Mm / MG), dim3(256), SMEM_BYTES, stream>>>(
      databf, sel, wbf, out);
}
